// FineRMoeSparseMoeBlock_27307402068613
// MI455X (gfx1250) — compile-verified, matrix operand order checked by reading, not run on hardware
//
#include <hip/hip_runtime.h>
#include <hip/hip_bf16.h>

// ---------------------------------------------------------------------------
// FineRMoE sparse MoE block for MI455X (gfx1250), wave32 + WMMA bf16.
// T=2048, H=1024, E=32, I=256, O=512, TOP_K=4, G_O=2.
// Grouped (token-gather) GEMMs: total gathered rows is exactly T*4 = 8192.
// All heavy GEMMs: fp32 global reads -> bf16 packed into LDS -> wmma bf16,
// fp32 accumulation. Block tile 64x128, 8 waves of 32x32 each, K-step 32.
// ---------------------------------------------------------------------------

#define T_TOK 2048
#define HDIM  1024
#define NEXP  32
#define IDIM  256
#define ODIM  512
#define CAP   2048            // worst-case tokens per expert
#define LDST  40              // padded LDS row stride (bf16 elems): 80B, 16B-aligned

typedef __bf16 bf16;
typedef __attribute__((ext_vector_type(16))) __bf16 v16bf;
typedef __attribute__((ext_vector_type(8)))  __bf16 v8bf;
typedef __attribute__((ext_vector_type(8)))  float  v8f;

// 16x32 bf16 WMMA operand fragment from a row-major LDS tile (row stride LDST).
// lane<16 -> K in {0..7,16..23}; lane>=16 -> K in {8..15,24..31}.
static __device__ __forceinline__ v16bf frag_ld(const bf16* tile, int row, int lane) {
  const bf16* p = tile + row * LDST + ((lane >> 4) << 3);
  v8bf lo = *(const v8bf*)(p);
  v8bf hi = *(const v8bf*)(p + 16);
  v16bf o;
#pragma unroll
  for (int i = 0; i < 8; ++i) { o[i] = lo[i]; o[i + 8] = hi[i]; }
  return o;
}

static __device__ __forceinline__ v8f vzero8() {
  v8f z = {0.f, 0.f, 0.f, 0.f, 0.f, 0.f, 0.f, 0.f};
  return z;
}

// pack two f32 -> one dword holding bf16(a) | bf16(b)<<16
static __device__ __forceinline__ unsigned pack2bf(float a, float b) {
  union { bf16 h[2]; unsigned u; } c;
  c.h[0] = (bf16)a; c.h[1] = (bf16)b;
  return c.u;
}

// convert 8 consecutive f32 -> v8bf (one ds_store_b128 worth)
static __device__ __forceinline__ v8bf cvt8(const float* s) {
  const float4 f0 = *(const float4*)(s);
  const float4 f1 = *(const float4*)(s + 4);
  v8bf v;
  v[0] = (bf16)f0.x; v[1] = (bf16)f0.y; v[2] = (bf16)f0.z; v[3] = (bf16)f0.w;
  v[4] = (bf16)f1.x; v[5] = (bf16)f1.y; v[6] = (bf16)f1.z; v[7] = (bf16)f1.w;
  return v;
}

static __device__ __forceinline__ float silu_fast(float g) {
  // g * sigmoid(g) with hardware v_rcp_f32 (no IEEE divide expansion)
  return g * __builtin_amdgcn_rcpf(1.f + __expf(-g));
}

// Transposed B-tile fill helper: thread owns k={2kp,2kp+1}, n=n0..n0+3.
// gRow0/gRow1 point at (k,n0) and (k+1,n0) in global fp32. Writes 4 dwords.
static __device__ __forceinline__ void fillB4(bf16* Bs, int n0, int kp,
                                              const float* gRow0, const float* gRow1) {
  const float4 r0 = *(const float4*)(gRow0);
  const float4 r1 = *(const float4*)(gRow1);
  *(unsigned*)&Bs[(n0 + 0) * LDST + 2 * kp] = pack2bf(r0.x, r1.x);
  *(unsigned*)&Bs[(n0 + 1) * LDST + 2 * kp] = pack2bf(r0.y, r1.y);
  *(unsigned*)&Bs[(n0 + 2) * LDST + 2 * kp] = pack2bf(r0.z, r1.z);
  *(unsigned*)&Bs[(n0 + 3) * LDST + 2 * kp] = pack2bf(r0.w, r1.w);
}

// ---------------------------------------------------------------------------
// K0: zero fp32 accumulator and per-expert counts.
// ---------------------------------------------------------------------------
__global__ void k0_zero(float* __restrict__ acc, int* __restrict__ counts) {
  int i = blockIdx.x * 256 + threadIdx.x;
  if (i < T_TOK * HDIM) acc[i] = 0.f;
  if (i < NEXP) counts[i] = 0;
}

// ---------------------------------------------------------------------------
// K1: router logits = x @ Wr ([2048,1024]@[1024,32]). One wave per token,
// lane = expert; writes into the router_logits section of d_out.
// ---------------------------------------------------------------------------
__global__ __launch_bounds__(256) void k1_router(const float* __restrict__ x,
                                                 const float* __restrict__ Wr,
                                                 float* __restrict__ logits) {
  int wave = threadIdx.x >> 5;
  int lane = threadIdx.x & 31;
  int t = blockIdx.x * 8 + wave;
  if (t >= T_TOK) return;
  const float* xr = x + (size_t)t * HDIM;
  float acc = 0.f;
#pragma unroll 8
  for (int h = 0; h < HDIM; ++h) acc = fmaf(xr[h], Wr[h * NEXP + lane], acc);
  logits[t * NEXP + lane] = acc;
}

// ---------------------------------------------------------------------------
// K2: FineRMoE routing. One thread per token; builds per-expert gather lists.
// ---------------------------------------------------------------------------
__global__ void k2_route(const float* __restrict__ logits,
                         int* __restrict__ counts,
                         int* __restrict__ tok,
                         float* __restrict__ wgt) {
  int t = blockIdx.x * blockDim.x + threadIdx.x;
  if (t >= T_TOK) return;
  float s[NEXP];
#pragma unroll
  for (int e = 0; e < NEXP; ++e) s[e] = logits[t * NEXP + e];

  float ssum[4];
  int top1[4], top2[4];
#pragma unroll
  for (int sh = 0; sh < 4; ++sh) {       // 4 shards of 8 experts
    float sum = 0.f, v1 = -3.0e38f, v2 = -3.0e38f;
    int b1 = 0, b2 = 0;
#pragma unroll
    for (int j = 0; j < 8; ++j) {
      float v = s[sh * 8 + j];
      sum += v;
      if (v > v1) { v2 = v1; b2 = b1; v1 = v; b1 = j; }
      else if (v > v2) { v2 = v; b2 = j; }
    }
    ssum[sh] = sum; top1[sh] = b1; top2[sh] = b2;
  }

  int ce[4]; float cs[4];
#pragma unroll
  for (int g = 0; g < 2; ++g) {          // winning shard per G_O group (first max)
    int sh = 2 * g + ((ssum[2 * g] >= ssum[2 * g + 1]) ? 0 : 1);
    ce[2 * g + 0] = sh * 8 + top1[sh];
    ce[2 * g + 1] = sh * 8 + top2[sh];
    cs[2 * g + 0] = s[ce[2 * g + 0]];
    cs[2 * g + 1] = s[ce[2 * g + 1]];
  }

  float m = fmaxf(fmaxf(cs[0], cs[1]), fmaxf(cs[2], cs[3]));
  float w[4], W = 0.f;
#pragma unroll
  for (int i = 0; i < 4; ++i) { w[i] = __expf(cs[i] - m); W += w[i]; }
  float inv = __builtin_amdgcn_rcpf(W);
#pragma unroll
  for (int i = 0; i < 4; ++i) {
    int e = ce[i];
    int slot = atomicAdd(&counts[e], 1);
    tok[e * CAP + slot] = t;
    wgt[e * CAP + slot] = w[i] * inv;
  }
}

// ---------------------------------------------------------------------------
// K3: grouped gate/up GEMM.  h = silu(Xe@Wg[e]) * (Xe@Wu[e]).
// Block tile M=64 x N=128 over I, K over H in steps of 32.
// 8 waves = 2(m) x 4(n); each wave owns 32x32 for gate AND up (8 accums).
// ---------------------------------------------------------------------------
__global__ __launch_bounds__(256) void k3_gateup(const float* __restrict__ x,
                                                 const float* __restrict__ Wg,
                                                 const float* __restrict__ Wu,
                                                 const int* __restrict__ counts,
                                                 const int* __restrict__ tok,
                                                 bf16* __restrict__ hbf) {
  int e  = blockIdx.x >> 5;
  int tt = blockIdx.x & 31;
  int nt = counts[e];
  if (tt * 64 >= nt) return;             // block-uniform exit: EXEC stays full
  int nb = blockIdx.y * 128;             // over I=256 (grid.y = 2)

  __shared__ bf16 As[64 * LDST];
  __shared__ bf16 Bg[128 * LDST];
  __shared__ bf16 Bu[128 * LDST];

  int tid = threadIdx.x;
  // A fill: row = tid>>2, cols (tid&3)*8..+7
  int arow = tid >> 2, acol = (tid & 3) << 3;
  int grow = tt * 64 + arow;
  int tix = (grow < nt) ? tok[e * CAP + grow] : 0;
  const float* xrow = x + (size_t)tix * HDIM;
  // B fill: k pair kp=tid&15, n groups (tid>>4)*4 and +64
  int kp = tid & 15;
  int ng = (tid >> 4) << 2;
  const float* wgp = Wg + (size_t)e * HDIM * IDIM + nb;
  const float* wup = Wu + (size_t)e * HDIM * IDIM + nb;

  int wv = tid >> 5, lane = tid & 31;
  int mw = (wv >> 2) << 5;               // 0,32
  int nw = (wv & 3) << 5;                // 0,32,64,96

  v8f ag[2][2], au[2][2];
#pragma unroll
  for (int i = 0; i < 2; ++i)
#pragma unroll
    for (int j = 0; j < 2; ++j) { ag[i][j] = vzero8(); au[i][j] = vzero8(); }

  for (int k = 0; k < HDIM; k += 32) {
    *(v8bf*)&As[arow * LDST + acol] = cvt8(xrow + k + acol);
    {
      const float* g0 = wgp + (size_t)(k + 2 * kp) * IDIM;
      const float* u0 = wup + (size_t)(k + 2 * kp) * IDIM;
      fillB4(Bg, ng,      kp, g0 + ng,      g0 + IDIM + ng);
      fillB4(Bg, ng + 64, kp, g0 + ng + 64, g0 + IDIM + ng + 64);
      fillB4(Bu, ng,      kp, u0 + ng,      u0 + IDIM + ng);
      fillB4(Bu, ng + 64, kp, u0 + ng + 64, u0 + IDIM + ng + 64);
      if (k + 32 < HDIM) {               // stream next weight rows toward L2
        __builtin_prefetch(g0 + 32 * IDIM, 0, 1);
        __builtin_prefetch(u0 + 32 * IDIM, 0, 1);
      }
    }
    __syncthreads();
    v16bf af[2], bgf[2], buf[2];
#pragma unroll
    for (int i = 0; i < 2; ++i) af[i] = frag_ld(As, mw + i * 16 + (lane & 15), lane);
#pragma unroll
    for (int j = 0; j < 2; ++j) {
      bgf[j] = frag_ld(Bg, nw + j * 16 + (lane & 15), lane);
      buf[j] = frag_ld(Bu, nw + j * 16 + (lane & 15), lane);
    }
#pragma unroll
    for (int i = 0; i < 2; ++i)
#pragma unroll
      for (int j = 0; j < 2; ++j) {
        ag[i][j] = __builtin_amdgcn_wmma_f32_16x16x32_bf16(false, af[i], false, bgf[j], (short)0, ag[i][j], false, false);
        au[i][j] = __builtin_amdgcn_wmma_f32_16x16x32_bf16(false, af[i], false, buf[j], (short)0, au[i][j], false, false);
      }
    __syncthreads();
  }

  int halfadd = (lane >> 4) << 3;
#pragma unroll
  for (int i = 0; i < 2; ++i)
#pragma unroll
    for (int j = 0; j < 2; ++j) {
      int colb = nb + nw + j * 16 + (lane & 15);
#pragma unroll
      for (int r = 0; r < 8; ++r) {
        int row = tt * 64 + mw + i * 16 + r + halfadd;
        if (row < nt) {
          float h = silu_fast(ag[i][j][r]) * au[i][j][r];
          hbf[((size_t)e * CAP + row) * IDIM + colb] = (bf16)h;
        }
      }
    }
}

// ---------------------------------------------------------------------------
// K4: grouped down GEMM + weighted scatter. y = h[e] @ Wd[e] ([*,256]@[256,512]),
// rows scaled by routing weight, atomically added into the e>>4 half of acc.
// Exactly two commutative float adds per output element => deterministic.
// Block tile 64x128 over O, K over I (8 steps).
// ---------------------------------------------------------------------------
__global__ __launch_bounds__(256) void k4_down(const bf16* __restrict__ hbf,
                                               const float* __restrict__ Wd,
                                               const int* __restrict__ counts,
                                               const int* __restrict__ tok,
                                               const float* __restrict__ wgt,
                                               float* __restrict__ acc) {
  int e  = blockIdx.x >> 5;
  int tt = blockIdx.x & 31;
  int nt = counts[e];
  if (tt * 64 >= nt) return;
  int nb = blockIdx.y * 128;             // over O=512 (grid.y = 4)

  __shared__ bf16 As[64 * LDST];
  __shared__ bf16 Bs[128 * LDST];

  int tid = threadIdx.x;
  int arow = tid >> 2, acol = (tid & 3) << 3;
  const bf16* hrow = hbf + ((size_t)e * CAP + tt * 64 + arow) * IDIM;
  int kp = tid & 15;
  int ng = (tid >> 4) << 2;
  const float* wdp = Wd + (size_t)e * IDIM * ODIM + nb;

  int wv = tid >> 5, lane = tid & 31;
  int mw = (wv >> 2) << 5;
  int nw = (wv & 3) << 5;

  v8f d[2][2];
#pragma unroll
  for (int i = 0; i < 2; ++i)
#pragma unroll
    for (int j = 0; j < 2; ++j) d[i][j] = vzero8();

  for (int k = 0; k < IDIM; k += 32) {
    *(v8bf*)&As[arow * LDST + acol] = *(const v8bf*)(hrow + k + acol);   // already bf16
    {
      const float* b0 = wdp + (size_t)(k + 2 * kp) * ODIM;
      fillB4(Bs, ng,      kp, b0 + ng,      b0 + ODIM + ng);
      fillB4(Bs, ng + 64, kp, b0 + ng + 64, b0 + ODIM + ng + 64);
      if (k + 32 < IDIM) __builtin_prefetch(b0 + 32 * ODIM, 0, 1);
    }
    __syncthreads();
    v16bf af[2], bf_[2];
#pragma unroll
    for (int i = 0; i < 2; ++i) af[i] = frag_ld(As, mw + i * 16 + (lane & 15), lane);
#pragma unroll
    for (int j = 0; j < 2; ++j) bf_[j] = frag_ld(Bs, nw + j * 16 + (lane & 15), lane);
#pragma unroll
    for (int i = 0; i < 2; ++i)
#pragma unroll
      for (int j = 0; j < 2; ++j)
        d[i][j] = __builtin_amdgcn_wmma_f32_16x16x32_bf16(false, af[i], false, bf_[j], (short)0, d[i][j], false, false);
    __syncthreads();
  }

  int halfadd = (lane >> 4) << 3;
  int gofs = (e >> 4) * ODIM;            // output-half offset in H
#pragma unroll
  for (int i = 0; i < 2; ++i)
#pragma unroll
    for (int r = 0; r < 8; ++r) {
      int row = tt * 64 + mw + i * 16 + r + halfadd;
      if (row < nt) {
        int t = tok[e * CAP + row];
        float w = wgt[e * CAP + row];
        float* dst = acc + (size_t)t * HDIM + gofs + nb + nw + (lane & 15);
#pragma unroll
        for (int j = 0; j < 2; ++j)
          atomicAdd(dst + j * 16, d[i][j][r] * w);
      }
    }
}

// ---------------------------------------------------------------------------
// K5: final concat projection: out = acc @ Wc ([2048,1024]@[1024,1024]).
// Block tile 64x128, K over H (32 steps).
// ---------------------------------------------------------------------------
__global__ __launch_bounds__(256) void k5_proj(const float* __restrict__ acc,
                                               const float* __restrict__ Wc,
                                               float* __restrict__ out) {
  int m0 = blockIdx.x * 64;              // 32 tiles over T
  int nb = blockIdx.y * 128;             // 8 tiles over H

  __shared__ bf16 As[64 * LDST];
  __shared__ bf16 Bs[128 * LDST];

  int tid = threadIdx.x;
  int arow = tid >> 2, acol = (tid & 3) << 3;
  const float* arowp = acc + (size_t)(m0 + arow) * HDIM;
  int kp = tid & 15;
  int ng = (tid >> 4) << 2;
  const float* wcp = Wc + nb;

  int wv = tid >> 5, lane = tid & 31;
  int mw = (wv >> 2) << 5;
  int nw = (wv & 3) << 5;

  v8f d[2][2];
#pragma unroll
  for (int i = 0; i < 2; ++i)
#pragma unroll
    for (int j = 0; j < 2; ++j) d[i][j] = vzero8();

  for (int k = 0; k < HDIM; k += 32) {
    *(v8bf*)&As[arow * LDST + acol] = cvt8(arowp + k + acol);
    {
      const float* b0 = wcp + (size_t)(k + 2 * kp) * HDIM;
      fillB4(Bs, ng,      kp, b0 + ng,      b0 + HDIM + ng);
      fillB4(Bs, ng + 64, kp, b0 + ng + 64, b0 + HDIM + ng + 64);
      if (k + 32 < HDIM) __builtin_prefetch(b0 + 32 * HDIM, 0, 1);
    }
    __syncthreads();
    v16bf af[2], bf_[2];
#pragma unroll
    for (int i = 0; i < 2; ++i) af[i] = frag_ld(As, mw + i * 16 + (lane & 15), lane);
#pragma unroll
    for (int j = 0; j < 2; ++j) bf_[j] = frag_ld(Bs, nw + j * 16 + (lane & 15), lane);
#pragma unroll
    for (int i = 0; i < 2; ++i)
#pragma unroll
      for (int j = 0; j < 2; ++j)
        d[i][j] = __builtin_amdgcn_wmma_f32_16x16x32_bf16(false, af[i], false, bf_[j], (short)0, d[i][j], false, false);
    __syncthreads();
  }

  int halfadd = (lane >> 4) << 3;
#pragma unroll
  for (int i = 0; i < 2; ++i)
#pragma unroll
    for (int j = 0; j < 2; ++j) {
      int col = nb + nw + j * 16 + (lane & 15);
#pragma unroll
      for (int r = 0; r < 8; ++r) {
        int row = m0 + mw + i * 16 + r + halfadd;
        out[(size_t)row * HDIM + col] = d[i][j][r];
      }
    }
}

// ---------------------------------------------------------------------------
// Host-side launch.
// inputs: 0=hidden_states[1,2048,1024] f32, 1=Wr[1024,32], 2=Wg[32,1024,256],
//         3=Wu[32,1024,256], 4=Wd[32,256,512], 5=Wc[1024,1024]
// d_out: out[2048*1024] f32 ++ router_logits[2048*32] f32
// ---------------------------------------------------------------------------
extern "C" void kernel_launch(void* const* d_in, const int* in_sizes, int n_in,
                              void* d_out, int out_size, void* d_ws, size_t ws_size,
                              hipStream_t stream) {
  const float* x  = (const float*)d_in[0];
  const float* Wr = (const float*)d_in[1];
  const float* Wg = (const float*)d_in[2];
  const float* Wu = (const float*)d_in[3];
  const float* Wd = (const float*)d_in[4];
  const float* Wc = (const float*)d_in[5];
  float* out    = (float*)d_out;
  float* logits = out + (size_t)T_TOK * HDIM;   // second reference output

  char* ws = (char*)d_ws;
  size_t off = 0;
  int* counts = (int*)(ws + off);   off += 256;
  int* tok    = (int*)(ws + off);   off += (size_t)NEXP * CAP * 4;
  float* wgt  = (float*)(ws + off); off += (size_t)NEXP * CAP * 4;
  off = (off + 255) & ~(size_t)255;
  bf16* hbf   = (bf16*)(ws + off);  off += (size_t)NEXP * CAP * IDIM * 2; // 32 MB
  float* acc  = (float*)(ws + off); off += (size_t)T_TOK * HDIM * 4;      // 8 MB
  (void)ws_size; (void)in_sizes; (void)n_in; (void)out_size;

  k0_zero  <<<(T_TOK * HDIM + 255) / 256, 256, 0, stream>>>(acc, counts);
  k1_router<<<T_TOK / 8, 256, 0, stream>>>(x, Wr, logits);
  k2_route <<<T_TOK / 256, 256, 0, stream>>>(logits, counts, tok, wgt);
  k3_gateup<<<dim3(NEXP * 32, IDIM / 128), 256, 0, stream>>>(x, Wg, Wu, counts, tok, hbf);
  k4_down  <<<dim3(NEXP * 32, ODIM / 128), 256, 0, stream>>>(hbf, Wd, counts, tok, wgt, acc);
  k5_proj  <<<dim3(T_TOK / 64, HDIM / 128), 256, 0, stream>>>(acc, Wc, out);
}
